// VectorQuantizer_584115552574
// MI455X (gfx1250) — matrix operand order repaired and checked by reading, not compiled, not run on hardware
//
#include <hip/hip_runtime.h>
#include <hip/hip_bf16.h>
#include <math.h>

// ---- problem constants (from reference) ----
#define KCODES 512
#define CDIM   64
#define HWDIM  4096            // 64*64
#define NROWS  131072          // 32*64*64
#define NELEM  8388608         // 32*64*64*64
#define ROWS_PER_BLOCK 128
#define STRH   72              // halves per padded row (144B: 16B aligned, conflict-free b128 frags)
#define CBH_BYTES (KCODES * STRH * 2)   // 73728 = 18 * 4096

typedef __attribute__((ext_vector_type(16))) _Float16 v16h;
typedef __attribute__((ext_vector_type(8)))  _Float16 v8h;
typedef __attribute__((ext_vector_type(2)))  _Float16 h2;
typedef __attribute__((ext_vector_type(8)))  float    v8f;

static __device__ __forceinline__ v16h cat16(v8h lo, v8h hi) {
    return __builtin_shufflevector(lo, hi, 0,1,2,3,4,5,6,7,8,9,10,11,12,13,14,15);
}

// -------------------------------------------------------------------------
// Kernel 0: one-time prep — codebook f32 -> f16 in padded LDS image layout,
//           plus fp32 ||c||^2 per code.
// -------------------------------------------------------------------------
__global__ __launch_bounds__(256) void vq_prep_kernel(
    const float* __restrict__ cb, float* __restrict__ cnormG,
    _Float16* __restrict__ cbH)
{
    const int r = blockIdx.x * 256 + threadIdx.x;   // 2 blocks -> 512 rows
    const float* row = cb + (size_t)r * CDIM;
    float s = 0.f;
    _Float16 tmp[CDIM];
    #pragma unroll
    for (int c = 0; c < CDIM; c += 4) {
        float4 v = *(const float4*)(row + c);
        s += v.x * v.x + v.y * v.y + v.z * v.z + v.w * v.w;
        tmp[c]     = (_Float16)v.x; tmp[c + 1] = (_Float16)v.y;
        tmp[c + 2] = (_Float16)v.z; tmp[c + 3] = (_Float16)v.w;
    }
    cnormG[r] = s;
    _Float16* dst = cbH + (size_t)r * STRH;         // padded image, pads never read
    #pragma unroll
    for (int c = 0; c < CDIM; c += 8)
        *(v8h*)(dst + c) = *(const v8h*)(tmp + c);
}

// -------------------------------------------------------------------------
// Kernel 1: per-row argmin over 512 codes via WMMA f16 (A=codes, B=z rows).
// Codebook image staged to LDS with CDNA5 async global->LDS copies,
// overlapped with the HBM-bound z transpose staging.
// -------------------------------------------------------------------------
__global__ __launch_bounds__(256) void vq_argmin_kernel(
    const float* __restrict__ z, const _Float16* __restrict__ cbH,
    const float* __restrict__ cnormG,
    int* __restrict__ idxOut, int* __restrict__ histG)
{
    __shared__ _Float16 cbLds[KCODES * STRH];          // 72 KB
    __shared__ _Float16 zLds[ROWS_PER_BLOCK * STRH];   // 18 KB
    __shared__ float    cnormLds[KCODES];              // 2 KB
    __shared__ int      histLds[KCODES];               // 2 KB

    const int t   = threadIdx.x;
    const int blk = blockIdx.x;                 // 1024 blocks: 32 per batch image
    const int b       = blk >> 5;               // HWDIM / ROWS_PER_BLOCK = 32
    const int base_hw = (blk & 31) * ROWS_PER_BLOCK;

    // ---- (a) kick off async L2->LDS copy of the prepacked f16 codebook ----
    {
        const unsigned ldsBase = (unsigned)(unsigned long long)(void*)&cbLds[0];
        const char* gBase = (const char*)cbH;
        #pragma unroll
        for (int i = 0; i < CBH_BYTES / 4096; ++i) {   // 18 rounds x 256 thr x 16B
            unsigned off = (unsigned)t * 16u + (unsigned)i * 4096u;
            unsigned laddr = ldsBase + off;
            unsigned long long gaddr = (unsigned long long)(gBase + off);
            asm volatile("global_load_async_to_lds_b128 %0, %1, off"
                         :: "v"(laddr), "v"(gaddr) : "memory");
        }
    }

    histLds[t] = 0; histLds[t + 256] = 0;
    cnormLds[t]       = cnormG[t];
    cnormLds[t + 256] = cnormG[t + 256];

    // ---- (b) stage 128 z rows (transpose [C][HW] -> LDS [row][C], f16) ----
    {
        const int hw_l = t & 127;
        const int ch   = t >> 7;                // 0/1: channel interleave, 2x coalesced segs
        const float* zb = z + (size_t)b * CDIM * HWDIM + base_hw;
        #pragma unroll 4
        for (int cc = 0; cc < 16; ++cc) {
            int c0 = cc * 4 + ch * 2;
            float a0 = zb[(size_t)c0 * HWDIM + hw_l];
            float a1 = zb[(size_t)(c0 + 1) * HWDIM + hw_l];
            h2 p = { (_Float16)a0, (_Float16)a1 };
            *(h2*)&zLds[hw_l * STRH + c0] = p;
        }
    }

    // ---- (c) wait for async codebook copy, then block barrier ----
    asm volatile("s_wait_asynccnt 0" ::: "memory");
    __syncthreads();

    // ---- WMMA sweep: each wave = 16 z rows vs all 512 codes ----
    const int lane = t & 31;
    const int wave = t >> 5;
    const int half = lane >> 4;                 // lane-half per ISA fragment layouts
    const int m    = lane & 15;

    // B fragments (z rows, loaded once): lane-half -> K {0..15}/{16..31} (+32 for k-chunk 1)
    const _Float16* zrow = &zLds[(16 * wave + m) * STRH];
    v16h bf0 = cat16(*(const v8h*)(zrow + half * 16),      *(const v8h*)(zrow + half * 16 + 8));
    v16h bf1 = cat16(*(const v8h*)(zrow + 32 + half * 16), *(const v8h*)(zrow + 32 + half * 16 + 8));

    float bestV = 3.4e38f;
    int   bestI = 0;

    for (int tile = 0; tile < 32; ++tile) {
        // A fragments (codes): lane-half -> K {0..15}/{16..31} (+32 for chunk 1)
        const _Float16* crow = &cbLds[(16 * tile + m) * STRH];
        v16h af0 = cat16(*(const v8h*)(crow + half * 16),      *(const v8h*)(crow + half * 16 + 8));
        v16h af1 = cat16(*(const v8h*)(crow + 32 + half * 16), *(const v8h*)(crow + 32 + half * 16 + 8));

        v8f acc = {};
        acc = __builtin_amdgcn_wmma_f32_16x16x32_f16(false, af0, false, bf0, (short)0, acc, false, false);
        acc = __builtin_amdgcn_wmma_f32_16x16x32_f16(false, af1, false, bf1, (short)0, acc, false, false);

        // D layout: VGPR r = code (16*tile + r + 8*half), col = lane&15 = z row
        const int cbase = tile * 16 + half * 8;
        #pragma unroll
        for (int r = 0; r < 8; ++r) {
            float score = cnormLds[cbase + r] - 2.0f * acc[r];   // ||c||^2 - 2 z.c
            if (score < bestV) { bestV = score; bestI = cbase + r; }
        }
    }

    // combine the two lane halves (same z row, code halves {..+0..7} vs {..+8..15})
    float ov = __shfl_xor(bestV, 16, 32);
    int   oi = __shfl_xor(bestI, 16, 32);
    if (ov < bestV || (ov == bestV && oi < bestI)) { bestV = ov; bestI = oi; }

    if (half == 0) {
        int rowG = blk * ROWS_PER_BLOCK + wave * 16 + m;
        idxOut[rowG] = bestI;
        atomicAdd(&histLds[bestI], 1);
    }
    __syncthreads();

    if (histLds[t])       atomicAdd(&histG[t],       histLds[t]);
    if (histLds[t + 256]) atomicAdd(&histG[t + 256], histLds[t + 256]);
}

// -------------------------------------------------------------------------
// Kernel 2: gather q = cb[idx] in [B,C,H,W] order + exact fp32 (q-z)^2 sums
// -------------------------------------------------------------------------
__global__ __launch_bounds__(256) void vq_gather_kernel(
    const float* __restrict__ z, const float* __restrict__ cb,
    const int* __restrict__ idx, float* __restrict__ out,
    float* __restrict__ partial)
{
    __shared__ float red[256];
    const int t = threadIdx.x;
    const size_t o4 = ((size_t)blockIdx.x * 256 + t) * 4;   // 4 consecutive elems, same (b,c)
    const int hw = (int)(o4 & (HWDIM - 1));
    const int c  = (int)((o4 >> 12) & 63);
    const int b  = (int)(o4 >> 18);
    const int n  = (b << 12) | hw;

    int4   iv = *(const int4*)(idx + n);
    float4 zv = *(const float4*)(z + o4);
    float q0 = cb[iv.x * CDIM + c];
    float q1 = cb[iv.y * CDIM + c];
    float q2 = cb[iv.z * CDIM + c];
    float q3 = cb[iv.w * CDIM + c];
    float4 qv = { q0, q1, q2, q3 };
    *(float4*)(out + o4) = qv;                               // quantized == q (STE)

    float d0 = q0 - zv.x, d1 = q1 - zv.y, d2 = q2 - zv.z, d3 = q3 - zv.w;
    red[t] = d0 * d0 + d1 * d1 + d2 * d2 + d3 * d3;
    __syncthreads();
    #pragma unroll
    for (int s = 128; s > 0; s >>= 1) {
        if (t < s) red[t] += red[t + s];
        __syncthreads();
    }
    if (t == 0) partial[blockIdx.x] = red[0];
}

// -------------------------------------------------------------------------
// Kernel 3: loss = 1.25 * mean((q-z)^2); perplexity from histogram
// -------------------------------------------------------------------------
__global__ __launch_bounds__(512) void vq_finalize_kernel(
    const int* __restrict__ hist, const float* __restrict__ partial,
    float* __restrict__ outScalars)
{
    __shared__ float redS[512];
    __shared__ float redE[512];
    const int t = threadIdx.x;

    float s = 0.f;
    for (int i = t; i < 8192; i += 512) s += partial[i];
    redS[t] = s;

    float p = (float)hist[t] * (1.0f / (float)NROWS);
    redE[t] = p * logf(p + 1e-10f);
    __syncthreads();
    #pragma unroll
    for (int k = 256; k > 0; k >>= 1) {
        if (t < k) { redS[t] += redS[t + k]; redE[t] += redE[t + k]; }
        __syncthreads();
    }
    if (t == 0) {
        outScalars[0] = 1.25f * redS[0] / (float)NELEM;  // q_loss + 0.25*e_loss
        outScalars[1] = expf(-redE[0]);                  // perplexity
    }
}

// -------------------------------------------------------------------------
extern "C" void kernel_launch(void* const* d_in, const int* in_sizes, int n_in,
                              void* d_out, int out_size, void* d_ws, size_t ws_size,
                              hipStream_t stream) {
    const float* z  = (const float*)d_in[0];   // [32,64,64,64] fp32
    const float* cb = (const float*)d_in[1];   // [512,64] fp32
    float* out = (float*)d_out;                // 8388608 quantized + loss + perplexity

    // workspace layout:
    //   hist    int[512]            @ 0
    //   partial float[8192]         @ 2048
    //   idx     int[131072]         @ 34816
    //   cnormG  float[512]          @ 559104
    //   cbH     f16[512*72] padded  @ 561152
    char* ws = (char*)d_ws;
    int*      hist    = (int*)ws;
    float*    partial = (float*)(ws + 2048);
    int*      idx     = (int*)(ws + 34816);
    float*    cnormG  = (float*)(ws + 559104);
    _Float16* cbH     = (_Float16*)(ws + 561152);

    hipMemsetAsync(d_ws, 0, 512 * sizeof(int) + 8192 * sizeof(float), stream);

    vq_prep_kernel    <<<2,                      256, 0, stream>>>(cb, cnormG, cbH);
    vq_argmin_kernel  <<<NROWS / ROWS_PER_BLOCK, 256, 0, stream>>>(z, cbH, cnormG, idx, hist);
    vq_gather_kernel  <<<NELEM / (256 * 4),      256, 0, stream>>>(z, cb, idx, out, partial);
    vq_finalize_kernel<<<1,                      512, 0, stream>>>(hist, partial, out + NELEM);
}
